// GraphFraudEnsemble_20323785244839
// MI455X (gfx1250) — matrix operand order repaired and checked by reading, not compiled, not run on hardware
//
#include <hip/hip_runtime.h>
#include <hip/hip_bf16.h>

// ---------------- problem constants (from reference setup_inputs) ----------------
#define NN     20000          // nodes
#define DIN    128            // input feature dim
#define HD2    256            // heads * ch = 4 * 64
#define NHEAD  4
#define CHD    64
#define NE     320000         // edges
#define NESL   (NE + NN)      // edges + self loops (GAT)
#define BN_EPS 1e-5f

typedef float  v2f   __attribute__((ext_vector_type(2)));
typedef float  v8f   __attribute__((ext_vector_type(8)));
typedef __bf16 v16bf __attribute__((ext_vector_type(16)));

#if defined(__gfx1250__) && __has_builtin(__builtin_amdgcn_wmma_f32_16x16x4_f32)
#define WMMA_PATH 2   // exact fp32 WMMA
#elif defined(__gfx1250__)
#define WMMA_PATH 1   // bf16 WMMA fallback (codegen-confirmed builtin)
#else
#define WMMA_PATH 0   // host-side parse only
#endif

// =====================================================================
// WMMA GEMM:  C[M,N] = A[M,K] @ W[N,K]^T (+bias) (+C)
// One wave -> 16x64 output strip (4 accumulators, A fragment reused 4x).
// M % 16 == 0, N % 64 == 0, K % 32 == 0 (K % 4 == 0 for fp32 path).
// =====================================================================
template<bool ADD_C, bool HAS_BIAS>
__global__ void k_gemm_wmma(const float* __restrict__ A, const float* __restrict__ W,
                            const float* __restrict__ bias, float* __restrict__ C,
                            int M, int N, int K)
{
    const int lane = threadIdx.x & 31;
    const int wave = blockIdx.x * (blockDim.x >> 5) + (threadIdx.x >> 5);
    const int colGroups = N >> 6;                 // 64-wide column groups
    const int tm = wave / colGroups;
    const int tg = wave - tm * colGroups;
    if (tm * 16 >= M) return;                     // wave-uniform exit (EXEC all-1 inside)

    const int half  = lane >> 4;                  // 0: lanes 0-15, 1: lanes 16-31
    const int mrow  = tm * 16 + (lane & 15);
    const int ncol0 = tg * 64 + (lane & 15);
    const float* __restrict__ arow = A + (size_t)mrow * K;

    v8f acc[4] = {};

#if WMMA_PATH == 2
    const int koff = half * 2;                    // A 16x4 f32: VGPR0=K{0|2}, VGPR1=K{1|3}
    for (int k = 0; k < K; k += 4) {
        v2f a; a.x = arow[k + koff]; a.y = arow[k + koff + 1];
#pragma unroll
        for (int t = 0; t < 4; ++t) {
            const float* wr = W + (size_t)(ncol0 + t * 16) * K + k + koff;
            v2f b; b.x = wr[0]; b.y = wr[1];      // B 4x16: row K striped across lanes
            acc[t] = __builtin_amdgcn_wmma_f32_16x16x4_f32(
                         false, a, false, b, (short)0, acc[t], false, false);
        }
    }
#elif WMMA_PATH == 1
    for (int k = 0; k < K; k += 32) {
        v16bf a;
#pragma unroll
        for (int j = 0; j < 16; ++j) {            // A 16x32 bf16 per-lane layout
            int v  = j >> 1;
            int kl = ((v >= 4) ? 16 : 0) + half * 8 + (v & 3) * 2 + (j & 1);
            a[j] = (__bf16)arow[k + kl];
        }
#pragma unroll
        for (int t = 0; t < 4; ++t) {
            const float* wr = W + (size_t)(ncol0 + t * 16) * K + k + half * 8;
            v16bf b;
#pragma unroll
            for (int j = 0; j < 16; ++j) b[j] = (__bf16)wr[(j < 8) ? j : j + 8];
            acc[t] = __builtin_amdgcn_wmma_f32_16x16x32_bf16(
                         false, a, false, b, (short)0, acc[t], false, false);
        }
    }
#else
    (void)arow; (void)ncol0;                      // host parse only; never executed
#endif

    const int rbase = half * 8;                   // C/D: lanes<16 -> M=r, lanes>=16 -> M=8+r
#pragma unroll
    for (int t = 0; t < 4; ++t) {
        const int col = ncol0 + t * 16;
        const float bv = HAS_BIAS ? bias[col] : 0.0f;
#pragma unroll
        for (int r = 0; r < 8; ++r) {
            size_t off = (size_t)(tm * 16 + rbase + r) * N + col;
            float val = acc[t][r] + bv;
            if (ADD_C) val += C[off];
            C[off] = val;
        }
    }
}

// =====================================================================
// Sparse / elementwise kernels
// =====================================================================
__global__ void k_deg(const int* __restrict__ dst, float* __restrict__ cnt, int E) {
    int i = blockIdx.x * blockDim.x + threadIdx.x;
    if (i < E) atomicAdd(&cnt[dst[i]], 1.0f);
}

__global__ void k_scatter_sum(const float* __restrict__ x, const int* __restrict__ src,
                              const int* __restrict__ dst, float* __restrict__ s, int E, int D) {
    long i = (long)blockIdx.x * blockDim.x + threadIdx.x;
    if (i >= (long)E * D) return;
    int e = (int)(i / D), d = (int)(i - (long)e * D);
    atomicAdd(&s[(long)dst[e] * D + d], x[(long)src[e] * D + d]);
}

__global__ void k_mean_div(float* __restrict__ s, const float* __restrict__ cnt, int Nn, int D) {
    long i = (long)blockIdx.x * blockDim.x + threadIdx.x;
    if (i >= (long)Nn * D) return;
    int n = (int)(i / D);
    s[i] = s[i] / fmaxf(cnt[n], 1.0f);
}

__global__ void k_bn_relu(const float* __restrict__ in, float* __restrict__ out,
                          const float* __restrict__ g, const float* __restrict__ b,
                          const float* __restrict__ rm, const float* __restrict__ rv,
                          const float* __restrict__ bias, int Nn, int D) {
    long i = (long)blockIdx.x * blockDim.x + threadIdx.x;
    if (i >= (long)Nn * D) return;
    int d = (int)(i % D);
    float v = in[i] + (bias ? bias[d] : 0.0f);
    v = (v - rm[d]) * rsqrtf(rv[d] + BN_EPS) * g[d] + b[d];
    out[i] = fmaxf(v, 0.0f);
}

// per-(node,head) attention coefficients:  a = sum_c h[n,h,c] * att[h,c]
__global__ void k_node_att(const float* __restrict__ h, const float* __restrict__ asw,
                           const float* __restrict__ adw, float* __restrict__ as_,
                           float* __restrict__ ad_, int Nn) {
    int i = blockIdx.x * blockDim.x + threadIdx.x;
    if (i >= Nn * NHEAD) return;
    int n = i / NHEAD, hh = i - n * NHEAD;
    const float* hp = h + (long)n * HD2 + hh * CHD;
    const float* aw = asw + hh * CHD;
    const float* dw = adw + hh * CHD;
    float s = 0.f, d2 = 0.f;
    for (int c = 0; c < CHD; ++c) { s += hp[c] * aw[c]; d2 += hp[c] * dw[c]; }
    as_[i] = s; ad_[i] = d2;
}

// GAT edge attention: ex = exp(leaky_relu(a_src[s]+a_dst[d])); den[d] += ex
__global__ void k_edge_exp_gat(const float* __restrict__ as_, const float* __restrict__ ad_,
                               const int* __restrict__ src, const int* __restrict__ dst,
                               float* __restrict__ ex, float* __restrict__ den, int E, int Esl) {
    long i = (long)blockIdx.x * blockDim.x + threadIdx.x;
    if (i >= (long)Esl * NHEAD) return;
    int e = (int)(i / NHEAD), hh = (int)(i - (long)e * NHEAD);
    int s = (e < E) ? src[e] : (e - E);           // implicit self-loop tail
    int d = (e < E) ? dst[e] : (e - E);
    float v = as_[s * NHEAD + hh] + ad_[d * NHEAD + hh];
    v = (v < 0.f) ? 0.2f * v : v;
    float ev = expf(v);
    ex[i] = ev;
    atomicAdd(&den[d * NHEAD + hh], ev);
}

// TransformerConv edge attention: ex = exp(q[d].k[s] / sqrt(C))
__global__ void k_edge_exp_tr(const float* __restrict__ q, const float* __restrict__ k,
                              const int* __restrict__ src, const int* __restrict__ dst,
                              float* __restrict__ ex, float* __restrict__ den, int E) {
    long i = (long)blockIdx.x * blockDim.x + threadIdx.x;
    if (i >= (long)E * NHEAD) return;
    int e = (int)(i / NHEAD), hh = (int)(i - (long)e * NHEAD);
    int s = src[e], d = dst[e];
    const float* qp = q + (long)d * HD2 + hh * CHD;
    const float* kp = k + (long)s * HD2 + hh * CHD;
    float acc = 0.f;
    for (int c = 0; c < CHD; ++c) acc += qp[c] * kp[c];
    acc *= 0.125f;                                // 1/sqrt(64)
    float ev = expf(acc);
    ex[i] = ev;
    atomicAdd(&den[d * NHEAD + hh], ev);
}

// out[d, :] += (ex/den[d]) * feat[s, :]   (per edge, 256 channels)
__global__ void k_edge_agg_256(const float* __restrict__ feat, const float* __restrict__ ex,
                               const float* __restrict__ den, const int* __restrict__ src,
                               const int* __restrict__ dst, float* __restrict__ out, int E, int Esl) {
    long i = (long)blockIdx.x * blockDim.x + threadIdx.x;
    if (i >= (long)Esl * HD2) return;
    int e = (int)(i / HD2), cf = (int)(i - (long)e * HD2);
    int hh = cf >> 6;
    int s = (e < E) ? src[e] : (e - E);
    int d = (e < E) ? dst[e] : (e - E);
    float w = ex[e * NHEAD + hh] / fmaxf(den[d * NHEAD + hh], 1e-16f);
    atomicAdd(&out[(long)d * HD2 + cf], w * feat[(long)s * HD2 + cf]);
}

// per-node dot product with one weight row (+scalar bias), optional accumulate
__global__ void k_dot_node(const float* __restrict__ feat, const float* __restrict__ w,
                           const float* __restrict__ b, float* __restrict__ out,
                           int Nn, int D, int add) {
    int n = blockIdx.x * blockDim.x + threadIdx.x;
    if (n >= Nn) return;
    const float* f = feat + (long)n * D;
    float acc = b ? b[0] : 0.0f;
    for (int i = 0; i < D; ++i) acc += f[i] * w[i];
    if (add) out[n] += acc; else out[n] = acc;
}

__global__ void k_gatl3_att(const float* __restrict__ t, const float* __restrict__ asw,
                            const float* __restrict__ adw, float* __restrict__ as_,
                            float* __restrict__ ad_, int Nn) {
    int n = blockIdx.x * blockDim.x + threadIdx.x;
    if (n >= Nn) return;
    as_[n] = t[n] * asw[0]; ad_[n] = t[n] * adw[0];
}

__global__ void k_edge_l3_gat(const float* __restrict__ as_, const float* __restrict__ ad_,
                              const int* __restrict__ src, const int* __restrict__ dst,
                              float* __restrict__ ex, float* __restrict__ den, int E, int Esl) {
    int e = blockIdx.x * blockDim.x + threadIdx.x;
    if (e >= Esl) return;
    int s = (e < E) ? src[e] : (e - E);
    int d = (e < E) ? dst[e] : (e - E);
    float v = as_[s] + ad_[d];
    v = (v < 0.f) ? 0.2f * v : v;
    float ev = expf(v);
    ex[e] = ev;
    atomicAdd(&den[d], ev);
}

__global__ void k_edge_l3_tr_exp(const float* __restrict__ qs, const float* __restrict__ ks,
                                 const int* __restrict__ src, const int* __restrict__ dst,
                                 float* __restrict__ ex, float* __restrict__ den, int E) {
    int e = blockIdx.x * blockDim.x + threadIdx.x;
    if (e >= E) return;
    float ev = expf(qs[dst[e]] * ks[src[e]]);     // ch=1 -> /sqrt(1)
    ex[e] = ev;
    atomicAdd(&den[dst[e]], ev);
}

__global__ void k_edge_l3_agg(const float* __restrict__ t, const float* __restrict__ ex,
                              const float* __restrict__ den, const int* __restrict__ src,
                              const int* __restrict__ dst, float* __restrict__ out, int E, int Esl) {
    int e = blockIdx.x * blockDim.x + threadIdx.x;
    if (e >= Esl) return;
    int s = (e < E) ? src[e] : (e - E);
    int d = (e < E) ? dst[e] : (e - E);
    atomicAdd(&out[d], (ex[e] / fmaxf(den[d], 1e-16f)) * t[s]);
}

__global__ void k_reduce_sum(const float* __restrict__ v, const float* __restrict__ biasScalar,
                             float* __restrict__ out, int Nn) {
    int n = blockIdx.x * blockDim.x + threadIdx.x;
    if (n >= Nn) return;
    atomicAdd(out, v[n] + (biasScalar ? biasScalar[0] : 0.0f));
}

// branch classifier: pooled scalar -> 64 hidden (relu) -> sigmoid scalar
__global__ void k_cls1(const float* __restrict__ sumPtr, float invN,
                       const float* __restrict__ w1, const float* __restrict__ b1,
                       const float* __restrict__ w2, const float* __restrict__ b2,
                       float* __restrict__ outSlot) {
    __shared__ float hsm[64];
    int i = threadIdx.x;
    float p = sumPtr[0] * invN;
    if (i < 64) hsm[i] = fmaxf(w1[i] * p + b1[i], 0.0f);
    __syncthreads();
    if (i == 0) {
        float acc = b2[0];
        for (int j = 0; j < 64; ++j) acc += w2[j] * hsm[j];
        outSlot[0] = 1.0f / (1.0f + expf(-acc));
    }
}

// final classifier over comb = [sage, gat, tr]
__global__ void k_final(const float* __restrict__ scal,
                        const float* __restrict__ w1, const float* __restrict__ b1,
                        const float* __restrict__ w2, const float* __restrict__ b2,
                        float* __restrict__ out) {
    __shared__ float hsm[64];
    int i = threadIdx.x;
    float c0 = scal[8], c1 = scal[9], c2 = scal[10];
    if (i < 64) hsm[i] = fmaxf(w1[i*3+0]*c0 + w1[i*3+1]*c1 + w1[i*3+2]*c2 + b1[i], 0.0f);
    __syncthreads();
    if (i == 0) {
        float acc = b2[0];
        for (int j = 0; j < 64; ++j) acc += w2[j] * hsm[j];
        out[0] = 1.0f / (1.0f + expf(-acc));
    }
}

// =====================================================================
// Host side: input resolution + launch sequence
// =====================================================================
enum {
    X_ = 0, EI_,
    S1WL, S1BL, S1WR, S2WL, S2BL, S2WR, S3WL, S3BL, S3WR,
    SB1G, SB1B, SB1RM, SB1RV, SB2G, SB2B, SB2RM, SB2RV,
    SCW1, SCB1, SCW2, SCB2,
    G1W, G1AS, G1AD, G1B, G2W, G2AS, G2AD, G2B, G3W, G3AS, G3AD, G3B,
    GB1G, GB1B, GB1RM, GB1RV, GB2G, GB2B, GB2RM, GB2RV,
    GCW1, GCB1, GCW2, GCB2,
    T1WQ, T1BQ, T1WK, T1BK, T1WV, T1BV, T1WS, T1BS,
    T2WQ, T2BQ, T2WK, T2BK, T2WV, T2BV, T2WS, T2BS,
    T3WQ, T3BQ, T3WK, T3BK, T3WV, T3BV, T3WS, T3BS,
    TB1G, TB1B, TB1RM, TB1RV, TB2G, TB2B, TB2RM, TB2RV,
    TCW1, TCB1, TCW2, TCB2,
    ENSW,
    CW1, CB1, CW2, CB2,
    NPARAMS
};

// canonical (insertion-order) slot -> d_in index when harness uses jax sorted-key flatten
static const int SORTED_MAP[NPARAMS] = {
    87, 0,
    43,42,44, 46,45,47, 49,48,50,
    31,30,32,33, 35,34,36,37,
    40,38,41,39,
    21,19,18,20, 25,23,22,24, 29,27,26,28,
    7,6,8,9, 11,10,12,13,
    16,14,17,15,
    68,64,67,63,70,66,69,65,
    76,72,75,71,78,74,77,73,
    84,80,83,79,86,82,85,81,
    52,51,53,54, 56,55,57,58,
    61,59,62,60,
    5,
    3,1,4,2
};

extern "C" void kernel_launch(void* const* d_in, const int* in_sizes, int n_in,
                              void* d_out, int out_size, void* d_ws, size_t ws_size,
                              hipStream_t stream)
{
    (void)n_in; (void)out_size; (void)ws_size;

    // ---- detect flatten order: sorted puts edge_index (640000 ints) first ----
    const bool sortedOrder = (in_sizes[0] == 2 * NE);
    int IDX[NPARAMS];
    for (int i = 0; i < NPARAMS; ++i) IDX[i] = sortedOrder ? SORTED_MAP[i] : i;

    auto Pf = [&](int c) -> const float* { return (const float*)d_in[IDX[c]]; };
    const int* EI  = (const int*)d_in[IDX[EI_]];
    const int* src = EI;
    const int* dst = EI + NE;
    const float* Xp = Pf(X_);

    // ---- workspace layout (floats) ----
    float* wsf = (float*)d_ws;
    size_t o = 0;
    auto alloc = [&](size_t n) { float* p = wsf + o; o += (n + 63) & ~(size_t)63; return p; };
    float* P0   = alloc((size_t)NN * HD2);     // h_prev (256)
    float* P1   = alloc((size_t)NN * HD2);     // q / gat hfeat
    float* P2   = alloc((size_t)NN * HD2);     // k
    float* P3   = alloc((size_t)NN * HD2);     // v
    float* P4   = alloc((size_t)NN * HD2);     // layer output / skip
    float* AGG  = alloc((size_t)NN * DIN);     // sage aggregate / mean
    float* T64A = alloc((size_t)NN * 64);
    float* T64B = alloc((size_t)NN * 64);
    float* CNT  = alloc(NN);
    float* A4A  = alloc((size_t)NN * NHEAD);
    float* A4B  = alloc((size_t)NN * NHEAD);
    float* DEN  = alloc((size_t)NN * NHEAD);
    float* EXB  = alloc((size_t)NESL * NHEAD);
    float* N1A  = alloc(NN);
    float* N1B  = alloc(NN);
    float* SCAL = alloc(256);

    auto grid = [](long n) { return dim3((unsigned)((n + 255) / 256)); };
    auto Z    = [&](void* p, size_t bytes) { hipMemsetAsync(p, 0, bytes, stream); };
    auto gemm = [&](const float* A, const float* W, const float* b, float* C,
                    int M, int Ncols, int K, int mode) {
        long waves = (long)(M / 16) * (Ncols / 64);
        unsigned blocks = (unsigned)((waves + 3) / 4);
        if (mode == 0)      k_gemm_wmma<false, true ><<<blocks, 128, 0, stream>>>(A, W, b, C, M, Ncols, K);
        else if (mode == 1) k_gemm_wmma<true , false><<<blocks, 128, 0, stream>>>(A, W, b, C, M, Ncols, K);
        else                k_gemm_wmma<false, false><<<blocks, 128, 0, stream>>>(A, W, b, C, M, Ncols, K);
    };

    Z(SCAL, 256 * sizeof(float));
    const float invN = 1.0f / (float)NN;

    // ===================== SAGE branch =====================
    Z(CNT, NN * sizeof(float));
    k_deg<<<grid(NE), 256, 0, stream>>>(dst, CNT, NE);
    // L1
    Z(AGG, (size_t)NN * DIN * sizeof(float));
    k_scatter_sum<<<grid((long)NE * DIN), 256, 0, stream>>>(Xp, src, dst, AGG, NE, DIN);
    k_mean_div<<<grid((long)NN * DIN), 256, 0, stream>>>(AGG, CNT, NN, DIN);
    gemm(AGG, Pf(S1WL), Pf(S1BL), T64A, NN, 64, DIN, 0);
    gemm(Xp,  Pf(S1WR), nullptr,  T64A, NN, 64, DIN, 1);
    k_bn_relu<<<grid((long)NN * 64), 256, 0, stream>>>(T64A, T64B, Pf(SB1G), Pf(SB1B), Pf(SB1RM), Pf(SB1RV), nullptr, NN, 64);
    // L2
    Z(AGG, (size_t)NN * 64 * sizeof(float));
    k_scatter_sum<<<grid((long)NE * 64), 256, 0, stream>>>(T64B, src, dst, AGG, NE, 64);
    k_mean_div<<<grid((long)NN * 64), 256, 0, stream>>>(AGG, CNT, NN, 64);
    gemm(AGG,  Pf(S2WL), Pf(S2BL), T64A, NN, 64, 64, 0);
    gemm(T64B, Pf(S2WR), nullptr,  T64A, NN, 64, 64, 1);
    k_bn_relu<<<grid((long)NN * 64), 256, 0, stream>>>(T64A, T64A, Pf(SB2G), Pf(SB2B), Pf(SB2RM), Pf(SB2RV), nullptr, NN, 64);
    // L3 (64 -> 1)
    Z(AGG, (size_t)NN * 64 * sizeof(float));
    k_scatter_sum<<<grid((long)NE * 64), 256, 0, stream>>>(T64A, src, dst, AGG, NE, 64);
    k_mean_div<<<grid((long)NN * 64), 256, 0, stream>>>(AGG, CNT, NN, 64);
    k_dot_node<<<grid(NN), 256, 0, stream>>>(AGG,  Pf(S3WL), Pf(S3BL), N1A, NN, 64, 0);
    k_dot_node<<<grid(NN), 256, 0, stream>>>(T64A, Pf(S3WR), nullptr,  N1A, NN, 64, 1);
    k_reduce_sum<<<grid(NN), 256, 0, stream>>>(N1A, nullptr, SCAL + 0, NN);
    k_cls1<<<1, 64, 0, stream>>>(SCAL + 0, invN, Pf(SCW1), Pf(SCB1), Pf(SCW2), Pf(SCB2), SCAL + 8);

    // ===================== GAT branch (self-loop edges) =====================
    // L1
    gemm(Xp, Pf(G1W), nullptr, P1, NN, HD2, DIN, 2);
    k_node_att<<<grid(NN * NHEAD), 256, 0, stream>>>(P1, Pf(G1AS), Pf(G1AD), A4A, A4B, NN);
    Z(DEN, (size_t)NN * NHEAD * sizeof(float));
    k_edge_exp_gat<<<grid((long)NESL * NHEAD), 256, 0, stream>>>(A4A, A4B, src, dst, EXB, DEN, NE, NESL);
    Z(P4, (size_t)NN * HD2 * sizeof(float));
    k_edge_agg_256<<<grid((long)NESL * HD2), 256, 0, stream>>>(P1, EXB, DEN, src, dst, P4, NE, NESL);
    k_bn_relu<<<grid((long)NN * HD2), 256, 0, stream>>>(P4, P0, Pf(GB1G), Pf(GB1B), Pf(GB1RM), Pf(GB1RV), Pf(G1B), NN, HD2);
    // L2
    gemm(P0, Pf(G2W), nullptr, P1, NN, HD2, HD2, 2);
    k_node_att<<<grid(NN * NHEAD), 256, 0, stream>>>(P1, Pf(G2AS), Pf(G2AD), A4A, A4B, NN);
    Z(DEN, (size_t)NN * NHEAD * sizeof(float));
    k_edge_exp_gat<<<grid((long)NESL * NHEAD), 256, 0, stream>>>(A4A, A4B, src, dst, EXB, DEN, NE, NESL);
    Z(P4, (size_t)NN * HD2 * sizeof(float));
    k_edge_agg_256<<<grid((long)NESL * HD2), 256, 0, stream>>>(P1, EXB, DEN, src, dst, P4, NE, NESL);
    k_bn_relu<<<grid((long)NN * HD2), 256, 0, stream>>>(P4, P0, Pf(GB2G), Pf(GB2B), Pf(GB2RM), Pf(GB2RV), Pf(G2B), NN, HD2);
    // L3 (heads=1, ch=1)
    k_dot_node<<<grid(NN), 256, 0, stream>>>(P0, Pf(G3W), nullptr, N1A, NN, HD2, 0);
    k_gatl3_att<<<grid(NN), 256, 0, stream>>>(N1A, Pf(G3AS), Pf(G3AD), A4A, A4B, NN);
    Z(DEN, NN * sizeof(float));
    k_edge_l3_gat<<<grid(NESL), 256, 0, stream>>>(A4A, A4B, src, dst, EXB, DEN, NE, NESL);
    Z(N1B, NN * sizeof(float));
    k_edge_l3_agg<<<grid(NESL), 256, 0, stream>>>(N1A, EXB, DEN, src, dst, N1B, NE, NESL);
    k_reduce_sum<<<grid(NN), 256, 0, stream>>>(N1B, Pf(G3B), SCAL + 1, NN);
    k_cls1<<<1, 64, 0, stream>>>(SCAL + 1, invN, Pf(GCW1), Pf(GCB1), Pf(GCW2), Pf(GCB2), SCAL + 9);

    // ===================== TransformerConv branch =====================
    // L1
    gemm(Xp, Pf(T1WQ), Pf(T1BQ), P1, NN, HD2, DIN, 0);
    gemm(Xp, Pf(T1WK), Pf(T1BK), P2, NN, HD2, DIN, 0);
    gemm(Xp, Pf(T1WV), Pf(T1BV), P3, NN, HD2, DIN, 0);
    gemm(Xp, Pf(T1WS), Pf(T1BS), P4, NN, HD2, DIN, 0);   // root skip pre-loaded into out
    Z(DEN, (size_t)NN * NHEAD * sizeof(float));
    k_edge_exp_tr<<<grid((long)NE * NHEAD), 256, 0, stream>>>(P1, P2, src, dst, EXB, DEN, NE);
    k_edge_agg_256<<<grid((long)NE * HD2), 256, 0, stream>>>(P3, EXB, DEN, src, dst, P4, NE, NE);
    k_bn_relu<<<grid((long)NN * HD2), 256, 0, stream>>>(P4, P0, Pf(TB1G), Pf(TB1B), Pf(TB1RM), Pf(TB1RV), nullptr, NN, HD2);
    // L2
    gemm(P0, Pf(T2WQ), Pf(T2BQ), P1, NN, HD2, HD2, 0);
    gemm(P0, Pf(T2WK), Pf(T2BK), P2, NN, HD2, HD2, 0);
    gemm(P0, Pf(T2WV), Pf(T2BV), P3, NN, HD2, HD2, 0);
    gemm(P0, Pf(T2WS), Pf(T2BS), P4, NN, HD2, HD2, 0);
    Z(DEN, (size_t)NN * NHEAD * sizeof(float));
    k_edge_exp_tr<<<grid((long)NE * NHEAD), 256, 0, stream>>>(P1, P2, src, dst, EXB, DEN, NE);
    k_edge_agg_256<<<grid((long)NE * HD2), 256, 0, stream>>>(P3, EXB, DEN, src, dst, P4, NE, NE);
    k_bn_relu<<<grid((long)NN * HD2), 256, 0, stream>>>(P4, P0, Pf(TB2G), Pf(TB2B), Pf(TB2RM), Pf(TB2RV), nullptr, NN, HD2);
    // L3 (heads=1, ch=1)
    k_dot_node<<<grid(NN), 256, 0, stream>>>(P0, Pf(T3WQ), Pf(T3BQ), A4A, NN, HD2, 0);
    k_dot_node<<<grid(NN), 256, 0, stream>>>(P0, Pf(T3WK), Pf(T3BK), A4B, NN, HD2, 0);
    k_dot_node<<<grid(NN), 256, 0, stream>>>(P0, Pf(T3WV), Pf(T3BV), N1A, NN, HD2, 0);
    k_dot_node<<<grid(NN), 256, 0, stream>>>(P0, Pf(T3WS), Pf(T3BS), N1B, NN, HD2, 0);  // skip init
    Z(DEN, NN * sizeof(float));
    k_edge_l3_tr_exp<<<grid(NE), 256, 0, stream>>>(A4A, A4B, src, dst, EXB, DEN, NE);
    k_edge_l3_agg<<<grid(NE), 256, 0, stream>>>(N1A, EXB, DEN, src, dst, N1B, NE, NE);
    k_reduce_sum<<<grid(NN), 256, 0, stream>>>(N1B, nullptr, SCAL + 2, NN);
    k_cls1<<<1, 64, 0, stream>>>(SCAL + 2, invN, Pf(TCW1), Pf(TCB1), Pf(TCW2), Pf(TCB2), SCAL + 10);

    // ===================== final ensemble classifier =====================
    k_final<<<1, 64, 0, stream>>>(SCAL, Pf(CW1), Pf(CB1), Pf(CW2), Pf(CB2), (float*)d_out);
}
